// RegionalAttentionMapGenerator_412316861061
// MI455X (gfx1250) — compile-verified, matrix operand order checked by reading, not run on hardware
//
#include <hip/hip_runtime.h>

typedef __attribute__((ext_vector_type(8))) int   v8i;
typedef __attribute__((ext_vector_type(8))) float v8f;
typedef __attribute__((ext_vector_type(4))) float v4f;

// Problem dimensions (fixed by setup_inputs: mask [64, 512, 896] fp32)
#define BDIM 64
#define HDIM 512
#define WDIM 896
#define W4   (WDIM / 4)                      // 224 float4 per row
#define F4_PER_IMG (HDIM * W4)               // 114688 float4 per image

// ---- pass 1 (reduction) geometry ----
#define SPLIT        8                       // blocks per image
#define RED_THREADS  256                     // 8 waves of 32
#define F4_PER_BLK   (F4_PER_IMG / SPLIT)    // 14336
#define F4_PER_ITER  (RED_THREADS * 8)       // 2048 (8 float4 per thread per iter)
#define N_ITERS      (F4_PER_BLK / F4_PER_ITER) // 7, exact

// ---- pass 2 (paint) geometry ----
#define WR_THREADS      256
#define F4_PER_WRBLK    (WR_THREADS * 4)          // 1024
#define WRBLK_PER_IMG   (F4_PER_IMG / F4_PER_WRBLK) // 112, exact

#define FP8_ONE 0x38  // E4M3 1.0

// Workspace layout: per image b, 5 ints at ws[b*5]: {cnt, rmin, rmax, cmin, cmax}
__global__ void rag_init_ws(int* __restrict__ ws) {
    int t = threadIdx.x;
    if (t < BDIM * 5) {
        int k = t % 5;
        int v = 0;
        if (k == 1) v = HDIM;        // rmin sentinel
        else if (k == 2) v = -1;     // rmax sentinel
        else if (k == 3) v = WDIM;   // cmin sentinel
        else if (k == 4) v = -1;     // cmax sentinel
        ws[t] = v;
    }
}

__global__ __launch_bounds__(RED_THREADS)
void rag_reduce(const float* __restrict__ mask, int* __restrict__ ws) {
    const unsigned bid = blockIdx.x;
    const unsigned b   = bid / SPLIT;
    const unsigned s   = bid % SPLIT;
    const unsigned t   = threadIdx.x;

    const v4f* img4 = (const v4f*)(mask + (size_t)b * HDIM * WDIM);
    const unsigned f4base = s * F4_PER_BLK;

    // All-ones B matrix for FP8 WMMA: every byte = 1.0f (E4M3 0x38)
    v8i bones;
#pragma unroll
    for (int i = 0; i < 8; ++i) bones[i] = 0x38383838;
    v8f acc = {};  // f32 16x16 accumulator; total(D) == 16 * count(flags), exact

    int rmin = HDIM, rmax = -1, cmin = WDIM, cmax = -1;

    for (int it = 0; it < N_ITERS; ++it) {
        // Stage all 8 b128 loads first: 8 outstanding loads per wave, no
        // exec-mask regions between them (memory-level parallelism).
        v4f v[8];
#pragma unroll
        for (int j = 0; j < 8; ++j) {
            const unsigned lin = f4base + it * F4_PER_ITER + j * RED_THREADS + t;
            v[j] = __builtin_nontemporal_load(&img4[lin]);  // streamed once: TH=NT
        }
        // Branchless processing (no saveexec between WMMAs / loads).
        v8i aflags;
#pragma unroll
        for (int j = 0; j < 8; ++j) {
            const unsigned lin = f4base + it * F4_PER_ITER + j * RED_THREADS + t;
            const int f0 = v[j].x > 0.5f;
            const int f1 = v[j].y > 0.5f;
            const int f2 = v[j].z > 0.5f;
            const int f3 = v[j].w > 0.5f;
            aflags[j] = (f0 ? FP8_ONE : 0) | (f1 ? (FP8_ONE << 8) : 0) |
                        (f2 ? (FP8_ONE << 16) : 0) | (f3 ? (FP8_ONE << 24) : 0);
            const int any = f0 | f1 | f2 | f3;
            const unsigned r  = lin / (unsigned)W4;          // absolute row
            const int      c0 = (int)(lin - r * (unsigned)W4) * 4;
            const int elo = f0 ? 0 : (f1 ? 1 : (f2 ? 2 : 3));
            const int ehi = f3 ? 3 : (f2 ? 2 : (f1 ? 1 : 0));
            rmin = min(rmin, any ? (int)r   : 0x7FFFFFFF);
            rmax = max(rmax, any ? (int)r   : -1);
            cmin = min(cmin, any ? c0 + elo : 0x7FFFFFFF);
            cmax = max(cmax, any ? c0 + ehi : -1);
        }
        // EXEC all-ones (branchless above). Counts 1024 pixels per wave:
        // D[m][n] = sum_k A[m][k] with B == 1.0.
        acc = __builtin_amdgcn_wmma_f32_16x16x64_fp8_fp8(
            aflags, bones, (short)0, acc, /*reuse_a=*/false, /*reuse_b=*/false);
    }

    __shared__ float s_cnt16;
    __shared__ int   s_rmin, s_rmax, s_cmin, s_cmax;
    if (t == 0) { s_cnt16 = 0.0f; s_rmin = HDIM; s_rmax = -1; s_cmin = WDIM; s_cmax = -1; }
    __syncthreads();

    // Per-lane partial of D; summing over all lanes of all waves = 16 * count.
    float lanesum = 0.0f;
#pragma unroll
    for (int i = 0; i < 8; ++i) lanesum += acc[i];
    atomicAdd(&s_cnt16, lanesum);
    if (rmax >= 0) {
        atomicMin(&s_rmin, rmin);
        atomicMax(&s_rmax, rmax);
        atomicMin(&s_cmin, cmin);
        atomicMax(&s_cmax, cmax);
    }
    __syncthreads();

    if (t == 0) {
        int* wsb = ws + b * 5;
        atomicAdd(&wsb[0], (int)(s_cnt16 * 0.0625f + 0.5f));  // /16, exact
        if (s_rmax >= 0) {
            atomicMin(&wsb[1], s_rmin);
            atomicMax(&wsb[2], s_rmax);
            atomicMin(&wsb[3], s_cmin);
            atomicMax(&wsb[4], s_cmax);
        }
    }
}

__global__ __launch_bounds__(WR_THREADS)
void rag_paint(const int* __restrict__ ws,
               const int* __restrict__ npts_thr,
               const int* __restrict__ dist_thr,
               float* __restrict__ out) {
    const unsigned bid = blockIdx.x;
    const unsigned b   = bid / WRBLK_PER_IMG;
    const unsigned s   = bid % WRBLK_PER_IMG;
    const unsigned t   = threadIdx.x;

    const int* wsb = ws + b * 5;
    const int  cnt  = wsb[0];
    const int  dist = dist_thr[0];
    const bool allones = cnt < npts_thr[0];
    const int  rlo = max(wsb[1] - dist, 0);
    const int  rhi = min(wsb[2] + dist, HDIM - 1);
    const int  clo = max(wsb[3] - dist, 0);
    const int  chi = min(wsb[4] + dist, WDIM - 1);

    v4f* out4 = (v4f*)(out + (size_t)b * HDIM * WDIM);
#pragma unroll
    for (int j = 0; j < 4; ++j) {
        const unsigned f4 = s * F4_PER_WRBLK + j * WR_THREADS + t;
        const unsigned r  = f4 / (unsigned)W4;
        const int      c0 = (int)(f4 - r * (unsigned)W4) * 4;
        const bool rin = ((int)r >= rlo) && ((int)r <= rhi);
        v4f v;
        v.x = (allones || (rin && (c0 + 0 >= clo) && (c0 + 0 <= chi))) ? 1.0f : 0.0f;
        v.y = (allones || (rin && (c0 + 1 >= clo) && (c0 + 1 <= chi))) ? 1.0f : 0.0f;
        v.z = (allones || (rin && (c0 + 2 >= clo) && (c0 + 2 <= chi))) ? 1.0f : 0.0f;
        v.w = (allones || (rin && (c0 + 3 >= clo) && (c0 + 3 <= chi))) ? 1.0f : 0.0f;
        __builtin_nontemporal_store(v, &out4[f4]);   // write-once stream: TH=NT
    }
}

extern "C" void kernel_launch(void* const* d_in, const int* in_sizes, int n_in,
                              void* d_out, int out_size, void* d_ws, size_t ws_size,
                              hipStream_t stream) {
    const float* mask     = (const float*)d_in[0];
    const int*   npts_thr = (const int*)d_in[1];
    const int*   dist_thr = (const int*)d_in[2];
    float*       out      = (float*)d_out;
    int*         ws       = (int*)d_ws;

    rag_init_ws<<<1, 512, 0, stream>>>(ws);
    rag_reduce<<<BDIM * SPLIT, RED_THREADS, 0, stream>>>(mask, ws);
    rag_paint<<<BDIM * WRBLK_PER_IMG, WR_THREADS, 0, stream>>>(ws, npts_thr, dist_thr, out);
}
